// LSTMBlock_46420006535299
// MI455X (gfx1250) — matrix operand order repaired and checked by reading, not compiled, not run on hardware
//
#include <hip/hip_runtime.h>

// ---------------------------------------------------------------------------
// LSTM block for MI455X (gfx1250): bf16 WMMA GEMM with fused gate epilogue.
//   pre = [x, h_prev] (16384x2048) @ Wc^T (4096x2048)  -> fused activations
// Double-buffered LDS staging via global_load_async_to_lds_b128 (ASYNCcnt).
// Each wave: 32 M-rows x (4 gates x 32 N) -> 16 WMMA / K-stage, 2x B reuse.
// Single-body K loop (XOR stage toggle) to keep accumulators register-pinned.
// ---------------------------------------------------------------------------

#define B_SZ 16384
#define I_SZ 1024
#define H_SZ 1024
#define K_SZ 2048     // I + H
#define NG   4        // gates: z, i, f, o

#define BM 256        // batch-rows per workgroup (8 waves x 32 rows)
#define BH 32         // hidden columns per workgroup (per gate)
#define BK 32         // K per stage (one WMMA K-depth)
#define PA 40         // padded LDS row stride (elements) for A tile
#define PB 40         // padded LDS row stride (elements) for B tile
#define NKT (K_SZ / BK)
#define STAGE_A (BM * PA)        // elements per A stage
#define STAGE_B (NG * BH * PB)   // elements per B stage

typedef __attribute__((ext_vector_type(16))) __bf16 v16bf;
typedef __attribute__((ext_vector_type(8)))  __bf16 v8bf;
typedef __attribute__((ext_vector_type(4)))  __bf16 v4bf;
typedef __attribute__((ext_vector_type(8)))  float  v8f;
typedef __attribute__((ext_vector_type(4)))  float  v4f;
typedef __attribute__((ext_vector_type(4)))  unsigned int v4u;
typedef __attribute__((ext_vector_type(4)))  int v4i;

#if defined(__HIP_DEVICE_COMPILE__) && \
    __has_builtin(__builtin_amdgcn_global_load_async_to_lds_b128) && \
    __has_builtin(__builtin_amdgcn_s_wait_asynccnt)
#define USE_ASYNC_LDS 1
typedef __attribute__((address_space(1))) v4i gv4i;   // global int4
typedef __attribute__((address_space(3))) v4i lv4i;   // LDS int4
#else
#define USE_ASYNC_LDS 0
#endif

// 16-byte global -> LDS copy (async on CDNA5, sync fallback otherwise)
__device__ __forceinline__ void copy16(const __bf16* gsrc, __bf16* ldst) {
#if USE_ASYNC_LDS
  __builtin_amdgcn_global_load_async_to_lds_b128((gv4i*)gsrc, (lv4i*)ldst, 0, 0);
#else
  *(v4u*)ldst = *(const v4u*)gsrc;
#endif
}

__device__ __forceinline__ void wait_async_le6() {
#if USE_ASYNC_LDS
  __builtin_amdgcn_s_wait_asynccnt(6);   // previous stage's 6 copies done
#endif
}
__device__ __forceinline__ void wait_async_le0() {
#if USE_ASYNC_LDS
  __builtin_amdgcn_s_wait_asynccnt(0);
#endif
}

// ---- pack [x | h_prev] -> bf16 Xc [B, 2048] ------------------------------
__global__ void lstm_cvt_x(const float* __restrict__ x,
                           const float* __restrict__ h_prev,
                           unsigned short* __restrict__ Xc_u) {
  __bf16* Xc = (__bf16*)Xc_u;
  int idx4 = blockIdx.x * blockDim.x + threadIdx.x;   // 4 elements per thread
  int b  = idx4 >> 9;                                 // / (2048/4)
  int kq = (idx4 & 511) << 2;
  const float* src = (kq < I_SZ) ? (x + (size_t)b * I_SZ + kq)
                                 : (h_prev + (size_t)b * H_SZ + (kq - I_SZ));
  v4f v = *(const v4f*)src;
  v4bf o;
  o[0] = (__bf16)v[0]; o[1] = (__bf16)v[1];
  o[2] = (__bf16)v[2]; o[3] = (__bf16)v[3];
  *(v4bf*)(Xc + (size_t)b * K_SZ + kq) = o;
}

// ---- pack [Wx | Rh] -> bf16 Wc [4096, 2048] (row n = (g,h), K contiguous) -
__global__ void lstm_cvt_w(const float* __restrict__ Wx,
                           const float* __restrict__ Rh,
                           unsigned short* __restrict__ Wc_u) {
  __bf16* Wc = (__bf16*)Wc_u;
  int idx4 = blockIdx.x * blockDim.x + threadIdx.x;
  int n  = idx4 >> 9;                // 0..4095
  int kq = (idx4 & 511) << 2;
  int g  = n >> 10;
  int hh = n & 1023;
  const float* src = (kq < I_SZ)
      ? (Wx + ((size_t)g * H_SZ + hh) * I_SZ + kq)
      : (Rh + ((size_t)g * H_SZ + hh) * H_SZ + (kq - I_SZ));
  v4f v = *(const v4f*)src;
  v4bf o;
  o[0] = (__bf16)v[0]; o[1] = (__bf16)v[1];
  o[2] = (__bf16)v[2]; o[3] = (__bf16)v[3];
  *(v4bf*)(Wc + (size_t)n * K_SZ + kq) = o;
}

__device__ __forceinline__ v16bf lds_frag(const __bf16* p0, const __bf16* p1) {
  v8bf lo = *(const v8bf*)p0;
  v8bf hi = *(const v8bf*)p1;
  return __builtin_shufflevector(lo, hi, 0, 1, 2, 3, 4, 5, 6, 7,
                                 8, 9, 10, 11, 12, 13, 14, 15);
}

// ---- main fused GEMM + LSTM epilogue --------------------------------------
__launch_bounds__(256)
__global__ void lstm_wmma_kernel(const unsigned short* __restrict__ Xc_u,
                                 const unsigned short* __restrict__ Wc_u,
                                 const float* __restrict__ c_prev,
                                 const float* __restrict__ bx,
                                 const float* __restrict__ bh,
                                 float* __restrict__ out) {
  const __bf16* Xc = (const __bf16*)Xc_u;
  const __bf16* Wc = (const __bf16*)Wc_u;

  __shared__ __bf16 ldsA[2 * STAGE_A];       // 2 x 20 KB
  __shared__ __bf16 ldsB[2 * STAGE_B];       // 2 x 10 KB

  const int tid  = threadIdx.x;
  const int lane = tid & 31;      // wave32
  const int wave = tid >> 5;      // 8 waves

  const int blockH = blockIdx.x * BH;
  const int blockM = blockIdx.y * BM;

  // ---- per-thread staging addresses (hoisted out of the K loop) ----------
  // A tile: 256 rows x 32 k -> 1024 16B chunks; thread handles 4 (row + j*64)
  const int rowC = tid >> 2;            // 0..63
  const int segC = (tid & 3) * 8;       // 0,8,16,24 (elements)
  const __bf16* gA[4];
  int oA[4];
#pragma unroll
  for (int j = 0; j < 4; ++j) {
    const int row = rowC + j * 64;
    gA[j] = Xc + (size_t)(blockM + row) * K_SZ + segC;
    oA[j] = row * PA + segC;
  }
  // B tiles: 4 gates x 32 n x 32 k -> 512 chunks; row = g*32+nn; 2 per thread
  const int gB = rowC >> 5, nnB = rowC & 31;
  const __bf16* gB0 = Wc + ((size_t)gB * H_SZ + blockH + nnB) * K_SZ + segC;
  const __bf16* gB1 = gB0 + (size_t)2 * H_SZ * K_SZ;   // gates g+2
  const int oB0 = rowC * PB + segC;
  const int oB1 = (rowC + 64) * PB + segC;

  v8f acc[NG][2][2];                    // [gate][n-tile][m-tile]
#pragma unroll
  for (int g = 0; g < NG; ++g)
#pragma unroll
    for (int t = 0; t < 2; ++t)
#pragma unroll
      for (int mt = 0; mt < 2; ++mt)
#pragma unroll
        for (int e = 0; e < 8; ++e) acc[g][t][mt][e] = 0.0f;

  const int lm = lane & 15;       // fragment row / column within 16
  const int lh = lane >> 4;       // half-select

  // LDS fragment base offsets (stage-relative, constant per thread)
  const int fA0 = (wave * 32 + lm) * PA + (lh << 4);           // m-tile 0
  const int fB0 = lm * PB + (lh << 4);                         // gate 0, t 0

  // ---- prologue: stage 0 -------------------------------------------------
#pragma unroll
  for (int j = 0; j < 4; ++j) copy16(gA[j], &ldsA[oA[j]]);
  copy16(gB0, &ldsB[oB0]);
  copy16(gB1, &ldsB[oB1]);

  int curA = 0, curB = 0;               // XOR-toggled stage offsets
#pragma clang loop unroll(disable)
  for (int kt = 0; kt < NKT; ++kt) {
    if (kt + 1 < NKT) {
      const int nxtA = curA ^ STAGE_A;
      const int nxtB = curB ^ STAGE_B;
      const int ko = (kt + 1) * BK;     // element offset along K
#pragma unroll
      for (int j = 0; j < 4; ++j) copy16(gA[j] + ko, &ldsA[nxtA + oA[j]]);
      copy16(gB0 + ko, &ldsB[nxtB + oB0]);
      copy16(gB1 + ko, &ldsB[nxtB + oB1]);
      wait_async_le6();                 // current stage's copies complete
    } else {
      wait_async_le0();
    }
    __syncthreads();

    // Two A fragments: this wave's rows [wave*32, wave*32+32)
    v16bf afr[2];
#pragma unroll
    for (int mt = 0; mt < 2; ++mt) {
      const __bf16* pa = &ldsA[curA + fA0 + mt * 16 * PA];
      afr[mt] = lds_frag(pa, pa + 8);
    }

    // 8 B fragments, each feeding 2 WMMAs (reused across both M-fragments)
#pragma unroll
    for (int g = 0; g < NG; ++g) {
#pragma unroll
      for (int t = 0; t < 2; ++t) {
        const __bf16* pb = &ldsB[curB + fB0 + (g * BH + t * 16) * PB];
        v16bf bfrag = lds_frag(pb, pb + 8);
        acc[g][t][0] = __builtin_amdgcn_wmma_f32_16x16x32_bf16(
            false, afr[0], false, bfrag, (short)0, acc[g][t][0], false, false);
        acc[g][t][1] = __builtin_amdgcn_wmma_f32_16x16x32_bf16(
            false, afr[1], false, bfrag, (short)0, acc[g][t][1], false, false);
      }
    }
    __syncthreads();   // all waves done with current stage before refill
    curA ^= STAGE_A;
    curB ^= STAGE_B;
  }

  // ---- fused LSTM epilogue -----------------------------------------------
  // C/D fragment: VGPR r, lane l -> M = r + (l>>4)*8, N = l&15
  float* out_h = out;
  float* out_c = out + (size_t)B_SZ * H_SZ;

#pragma unroll
  for (int t = 0; t < 2; ++t) {
    const int h = blockH + t * 16 + lm;
    const float bz = bx[0 * H_SZ + h] + bh[0 * H_SZ + h];
    const float bi = bx[1 * H_SZ + h] + bh[1 * H_SZ + h];
    const float bf = bx[2 * H_SZ + h] + bh[2 * H_SZ + h];
    const float bo = bx[3 * H_SZ + h] + bh[3 * H_SZ + h];
#pragma unroll
    for (int mt = 0; mt < 2; ++mt) {
      const int m_base = blockM + wave * 32 + mt * 16 + (lh << 3);
#pragma unroll
      for (int r = 0; r < 8; ++r) {
        const int b = m_base + r;
        const size_t off = (size_t)b * H_SZ + h;
        const float cp = c_prev[off];
        const float z  = tanhf(acc[0][t][mt][r] + bz);
        const float ig = 1.0f / (1.0f + __expf(-(acc[1][t][mt][r] + bi)));
        const float fg = 1.0f / (1.0f + __expf(-(acc[2][t][mt][r] + bf)));
        const float og = 1.0f / (1.0f + __expf(-(acc[3][t][mt][r] + bo)));
        const float cn = fg * cp + ig * z;
        out_c[off] = cn;
        out_h[off] = og * tanhf(cn);
      }
    }
  }
}

// ---------------------------------------------------------------------------
extern "C" void kernel_launch(void* const* d_in, const int* in_sizes, int n_in,
                              void* d_out, int out_size, void* d_ws, size_t ws_size,
                              hipStream_t stream) {
  (void)in_sizes; (void)n_in; (void)out_size; (void)ws_size;

  const float* x      = (const float*)d_in[0];
  const float* h_prev = (const float*)d_in[1];
  const float* c_prev = (const float*)d_in[2];
  const float* Wx     = (const float*)d_in[3];
  const float* bx     = (const float*)d_in[4];
  const float* Rh     = (const float*)d_in[5];
  const float* bh     = (const float*)d_in[6];

  unsigned short* Xc = (unsigned short*)d_ws;                       // 64 MB bf16
  unsigned short* Wc = Xc + (size_t)B_SZ * K_SZ;                    // +16 MB bf16

  {
    const int threads = (B_SZ * K_SZ) / 4;       // 4 elems per thread
    lstm_cvt_x<<<threads / 256, 256, 0, stream>>>(x, h_prev, Xc);
  }
  {
    const int threads = (NG * H_SZ * K_SZ) / 4;
    lstm_cvt_w<<<threads / 256, 256, 0, stream>>>(Wx, Rh, Wc);
  }
  {
    dim3 grid(H_SZ / BH, B_SZ / BM);             // 32 x 64 workgroups
    lstm_wmma_kernel<<<grid, 256, 0, stream>>>(Xc, Wc, c_prev, bx, bh,
                                               (float*)d_out);
  }
}
